// SelfAttention_36120674959815
// MI455X (gfx1250) — matrix operand order repaired and checked by reading, not compiled, not run on hardware
//
#include <hip/hip_runtime.h>

#define USE_ASYNC_LDS 1

typedef __bf16 bf16;
typedef __attribute__((ext_vector_type(16))) __bf16 v16bf;
typedef __attribute__((ext_vector_type(8)))  float  v8f;

constexpr int kB  = 2;
constexpr int kT  = 2048;
constexpr int kE  = 1024;
constexpr int kH  = 16;
constexpr int kD  = 128;
constexpr int kHD = 2048;   // N_HEAD * D
constexpr int kM  = 4096;   // B * T

__device__ __forceinline__ bf16 f2bf(float x) { return (bf16)x; }

__device__ __forceinline__ v8f zero8() {
  v8f z;
  #pragma unroll
  for (int i = 0; i < 8; ++i) z[i] = 0.0f;
  return z;
}

__device__ __forceinline__ v8f wmma_bf16(v16bf a, v16bf b, v8f c) {
  return __builtin_amdgcn_wmma_f32_16x16x32_bf16(false, a, false, b, (short)0, c,
                                                 false, false);
}

__device__ __forceinline__ v16bf load16bf(const bf16* p) {
  return *(const v16bf*)p;
}

#if USE_ASYNC_LDS
// CDNA5: async DMA 16B global -> LDS, tracked by ASYNCcnt (GV addressing)
__device__ __forceinline__ void async_copy16(void* lds, const bf16* g) {
  unsigned lds_off = (unsigned)(size_t)lds;            // low 32b = LDS offset
  unsigned long long ga = (unsigned long long)(size_t)g;
  asm volatile("global_load_async_to_lds_b128 %0, %1, off"
               :: "v"(lds_off), "v"(ga) : "memory");
}
__device__ __forceinline__ void async_wait0() {
  asm volatile("s_wait_asynccnt 0" ::: "memory");
}
#else
__device__ __forceinline__ void async_copy16(void* lds, const bf16* g) {
  *(uint4*)lds = *(const uint4*)g;
}
__device__ __forceinline__ void async_wait0() {}
#endif

// -------------------------------------------------------------------------
// One-time conversion kernels (feed the async DMA path with bf16 data)
// -------------------------------------------------------------------------
__global__ __launch_bounds__(256) void cvt_x_kernel(
    const float* __restrict__ in, bf16* __restrict__ out) {
  int c = (blockIdx.x * 256 + threadIdx.x) * 4;
  float4 v = *(const float4*)&in[c];
  union { bf16 h[4]; uint2 u; } t;
  t.h[0] = f2bf(v.x); t.h[1] = f2bf(v.y); t.h[2] = f2bf(v.z); t.h[3] = f2bf(v.w);
  *(uint2*)&out[c] = t.u;
}

// W[K][N] f32 -> WT[N][K] bf16, scaled (scale folds 1/sqrt(D) into Wq)
__global__ __launch_bounds__(256) void cvt_wT_kernel(
    const float* __restrict__ W, bf16* __restrict__ WT, int K, int N,
    float scale) {
  int k = blockIdx.y;
  int c = blockIdx.x * 1024 + threadIdx.x * 4;
  float4 v = *(const float4*)&W[(size_t)k * N + c];
  WT[(size_t)(c + 0) * K + k] = f2bf(v.x * scale);
  WT[(size_t)(c + 1) * K + k] = f2bf(v.y * scale);
  WT[(size_t)(c + 2) * K + k] = f2bf(v.z * scale);
  WT[(size_t)(c + 3) * K + k] = f2bf(v.w * scale);
}

// -------------------------------------------------------------------------
// GEMM: C[m,n] = A[m,:] * WT[n,:]^T + biasScale*bias[n]
// A bf16 [M][K], WT bf16 [N][K]. Async double-buffered LDS staging.
// Block tile 64x128, k-tile 64, 8 waves (2Mx4N), wave 32x32.
// VMODE: 0 -> dst [b,h,t,d] bf16 ; 1 -> dst [b,h,d,t] bf16 (packed stores)
// -------------------------------------------------------------------------
template <int VMODE, int KTOT>
__global__ __launch_bounds__(256) void gemm_bf16_kernel(
    const bf16* __restrict__ A, const bf16* __restrict__ WT,
    const float* __restrict__ bias, float biasScale, bf16* __restrict__ dst) {
  __shared__ bf16 As[2][64][72];     // [m][k]
  __shared__ bf16 BsT[2][128][72];   // [n][k]

  const int tid  = threadIdx.x;
  const int lane = tid & 31;
  const int wid  = tid >> 5;
  const int hi   = lane >> 4;
  const int ln   = lane & 15;
  const int waveM = wid >> 2;
  const int waveN = wid & 3;
  const int mBase = blockIdx.y * 64;
  const int nBase = blockIdx.x * 128;

  auto stage = [&](int buf, int k0) {
    #pragma unroll
    for (int i = 0; i < 2; ++i) {           // A: 512 x 16B
      int c = tid + i * 256;
      int row = c >> 3, kc = (c & 7) << 3;
      async_copy16(&As[buf][row][kc], &A[(size_t)(mBase + row) * KTOT + k0 + kc]);
    }
    #pragma unroll
    for (int i = 0; i < 4; ++i) {           // B: 1024 x 16B
      int c = tid + i * 256;
      int nb = c >> 3, kc = (c & 7) << 3;
      async_copy16(&BsT[buf][nb][kc], &WT[(size_t)(nBase + nb) * KTOT + k0 + kc]);
    }
  };

  v8f acc[2][2];
  #pragma unroll
  for (int i = 0; i < 2; ++i)
    #pragma unroll
    for (int j = 0; j < 2; ++j) acc[i][j] = zero8();

  constexpr int NK = KTOT / 64;
  stage(0, 0);
  for (int ki = 0; ki < NK; ++ki) {
    int cur = ki & 1;
    async_wait0();
    __syncthreads();
    if (ki + 1 < NK) stage(cur ^ 1, (ki + 1) * 64);

    #pragma unroll
    for (int ks = 0; ks < 2; ++ks) {
      v16bf a[2], b[2];
      #pragma unroll
      for (int mt = 0; mt < 2; ++mt) {
        int row = waveM * 32 + mt * 16 + ln;
        #pragma unroll
        for (int e = 0; e < 16; ++e) {
          int k = ks * 32 + (hi << 4) + e;
          a[mt][e] = As[cur][row][k];
        }
      }
      #pragma unroll
      for (int nt = 0; nt < 2; ++nt) {
        int col = waveN * 32 + nt * 16 + ln;
        b[nt] = load16bf(&BsT[cur][col][ks * 32 + (hi << 4)]);
      }
      #pragma unroll
      for (int mt = 0; mt < 2; ++mt)
        #pragma unroll
        for (int nt = 0; nt < 2; ++nt)
          acc[mt][nt] = wmma_bf16(a[mt], b[nt], acc[mt][nt]);
    }
  }

  #pragma unroll
  for (int mt = 0; mt < 2; ++mt) {
    #pragma unroll
    for (int nt = 0; nt < 2; ++nt) {
      int row0 = mBase + waveM * 32 + mt * 16 + hi * 8;
      int col  = nBase + waveN * 32 + nt * 16 + ln;
      float bv = biasScale * bias[col];
      int bb = row0 >> 11, tt0 = row0 & (kT - 1);
      int hh = col >> 7,  dd = col & (kD - 1);
      if (VMODE == 0) {
        bf16* p = dst + (((size_t)bb * kH + hh) * kT + tt0) * kD + dd;
        #pragma unroll
        for (int r = 0; r < 8; ++r) p[(size_t)r * kD] = f2bf(acc[mt][nt][r] + bv);
      } else {
        union { bf16 h[8]; uint4 u; } pk;
        #pragma unroll
        for (int r = 0; r < 8; ++r) pk.h[r] = f2bf(acc[mt][nt][r] + bv);
        *(uint4*)(dst + (((size_t)bb * kH + hh) * kD + dd) * kT + tt0) = pk.u;
      }
    }
  }
}

// -------------------------------------------------------------------------
// Output projection variant: f32 result to d_out.
// -------------------------------------------------------------------------
__global__ __launch_bounds__(256) void out_proj_kernel(
    const bf16* __restrict__ A, const bf16* __restrict__ WT,
    const float* __restrict__ bias, float* __restrict__ out) {
  __shared__ bf16 As[2][64][72];
  __shared__ bf16 BsT[2][128][72];

  const int tid  = threadIdx.x;
  const int lane = tid & 31;
  const int wid  = tid >> 5;
  const int hi   = lane >> 4;
  const int ln   = lane & 15;
  const int waveM = wid >> 2;
  const int waveN = wid & 3;
  const int mBase = blockIdx.y * 64;
  const int nBase = blockIdx.x * 128;

  auto stage = [&](int buf, int k0) {
    #pragma unroll
    for (int i = 0; i < 2; ++i) {
      int c = tid + i * 256;
      int row = c >> 3, kc = (c & 7) << 3;
      async_copy16(&As[buf][row][kc], &A[(size_t)(mBase + row) * kHD + k0 + kc]);
    }
    #pragma unroll
    for (int i = 0; i < 4; ++i) {
      int c = tid + i * 256;
      int nb = c >> 3, kc = (c & 7) << 3;
      async_copy16(&BsT[buf][nb][kc], &WT[(size_t)(nBase + nb) * kHD + k0 + kc]);
    }
  };

  v8f acc[2][2];
  #pragma unroll
  for (int i = 0; i < 2; ++i)
    #pragma unroll
    for (int j = 0; j < 2; ++j) acc[i][j] = zero8();

  constexpr int NK = kHD / 64;
  stage(0, 0);
  for (int ki = 0; ki < NK; ++ki) {
    int cur = ki & 1;
    async_wait0();
    __syncthreads();
    if (ki + 1 < NK) stage(cur ^ 1, (ki + 1) * 64);

    #pragma unroll
    for (int ks = 0; ks < 2; ++ks) {
      v16bf a[2], b[2];
      #pragma unroll
      for (int mt = 0; mt < 2; ++mt) {
        int row = waveM * 32 + mt * 16 + ln;
        #pragma unroll
        for (int e = 0; e < 16; ++e) {
          int k = ks * 32 + (hi << 4) + e;
          a[mt][e] = As[cur][row][k];
        }
      }
      #pragma unroll
      for (int nt = 0; nt < 2; ++nt) {
        int col = waveN * 32 + nt * 16 + ln;
        b[nt] = load16bf(&BsT[cur][col][ks * 32 + (hi << 4)]);
      }
      #pragma unroll
      for (int mt = 0; mt < 2; ++mt)
        #pragma unroll
        for (int nt = 0; nt < 2; ++nt)
          acc[mt][nt] = wmma_bf16(a[mt], b[nt], acc[mt][nt]);
    }
  }

  #pragma unroll
  for (int mt = 0; mt < 2; ++mt) {
    #pragma unroll
    for (int nt = 0; nt < 2; ++nt) {
      int row0 = mBase + waveM * 32 + mt * 16 + hi * 8;
      int col  = nBase + waveN * 32 + nt * 16 + ln;
      float bv = bias[col];
      float* p = out + (size_t)row0 * kE + col;
      #pragma unroll
      for (int r = 0; r < 8; ++r) p[(size_t)r * kE] = acc[mt][nt][r] + bv;
    }
  }
}

// -------------------------------------------------------------------------
// Causal flash attention per (b,h). 256 threads (8 waves), 128 query rows
// per block (16 per wave). Key blocks of 64, async double-buffered K/V.
// Q,K in [b,h,t,d] (Q pre-scaled by 1/sqrt(D)); V in [b,h,d,t].
// -------------------------------------------------------------------------
__global__ __launch_bounds__(256) void attn_kernel(
    const bf16* __restrict__ Q, const bf16* __restrict__ K,
    const bf16* __restrict__ V, bf16* __restrict__ O) {
  __shared__ bf16 Ks[2][64][136];    // [key][d]
  __shared__ bf16 VsT[2][128][72];   // [d][key]
  __shared__ bf16 Ps[8][16][72];     // per-wave P tile [qrow][key]

  const int tid  = threadIdx.x;
  const int lane = tid & 31;
  const int wid  = tid >> 5;
  const int hi   = lane >> 4;
  const int ln   = lane & 15;
  const int qblk = blockIdx.x;
  const int h    = blockIdx.y;
  const int b    = blockIdx.z;
  const size_t head = ((size_t)b * kH + h) * (size_t)kT * kD;
  const bf16* Qh = Q + head;
  const bf16* Kh = K + head;
  const bf16* Vh = V + head;          // [d][t]
  const int qw = qblk * 128 + wid * 16;

  auto stageKV = [&](int buf, int kb) {
    #pragma unroll
    for (int t = 0; t < 4; ++t) {       // K tile: 1024 x 16B
      int c  = tid + t * 256;
      int r  = c >> 4;
      int cp = (c & 15) << 3;
      async_copy16(&Ks[buf][r][cp], &Kh[(size_t)(kb * 64 + r) * kD + cp]);
    }
    #pragma unroll
    for (int t = 0; t < 4; ++t) {       // V tile: 1024 x 16B
      int c  = tid + t * 256;
      int d  = c >> 3;
      int kp = (c & 7) << 3;
      async_copy16(&VsT[buf][d][kp], &Vh[(size_t)d * kT + kb * 64 + kp]);
    }
  };

  // Q fragments (A layout), 4 k-steps of 32 (Q already scaled by 1/sqrt(D))
  v16bf qf[4];
  #pragma unroll
  for (int ks = 0; ks < 4; ++ks)
    #pragma unroll
    for (int e = 0; e < 16; ++e) {
      int d = ks * 32 + (hi << 4) + e;
      qf[ks][e] = Qh[(size_t)(qw + ln) * kD + d];
    }

  v8f o[8];
  float mrow[8], lrow[8];
  #pragma unroll
  for (int dt = 0; dt < 8; ++dt) o[dt] = zero8();
  #pragma unroll
  for (int r = 0; r < 8; ++r) { mrow[r] = -3.0e38f; lrow[r] = 0.0f; }

  const int NKB = 2 * qblk + 2;       // key blocks covering causal range
  stageKV(0, 0);
  for (int kb = 0; kb < NKB; ++kb) {
    int cur = kb & 1;
    async_wait0();
    __syncthreads();
    if (kb + 1 < NKB) stageKV(cur ^ 1, kb + 1);

    const bool active = (kb * 64) <= (qw + 15);   // wave-uniform
    if (active) {
      // S = Q * K^T  (16 x 64)
      v8f s[4];
      #pragma unroll
      for (int j = 0; j < 4; ++j) s[j] = zero8();
      #pragma unroll
      for (int j = 0; j < 4; ++j)
        #pragma unroll
        for (int ks = 0; ks < 4; ++ks) {
          v16bf kf = load16bf(&Ks[cur][j * 16 + ln][ks * 32 + hi * 16]);
          s[j] = wmma_bf16(qf[ks], kf, s[j]);
        }

      // causal mask only on diagonal blocks (wave-uniform branch)
      if (kb * 64 + 63 > qw) {
        #pragma unroll
        for (int r = 0; r < 8; ++r) {
          int qr = qw + r + hi * 8;
          #pragma unroll
          for (int j = 0; j < 4; ++j) {
            int key = kb * 64 + j * 16 + ln;
            if (key > qr) s[j][r] = -3.0e38f;
          }
        }
      }

      // online softmax
      float mnew[8];
      #pragma unroll
      for (int r = 0; r < 8; ++r) {
        float mx = fmaxf(fmaxf(s[0][r], s[1][r]), fmaxf(s[2][r], s[3][r]));
        #pragma unroll
        for (int off = 1; off < 16; off <<= 1)
          mx = fmaxf(mx, __shfl_xor(mx, off, 32));
        mnew[r] = fmaxf(mrow[r], mx);
      }
      #pragma unroll
      for (int r = 0; r < 8; ++r) {
        float alpha = __expf(mrow[r] - mnew[r]);
        lrow[r] *= alpha;
        #pragma unroll
        for (int dt = 0; dt < 8; ++dt) o[dt][r] *= alpha;
        float rs = 0.0f;
        #pragma unroll
        for (int j = 0; j < 4; ++j) {
          float p = __expf(s[j][r] - mnew[r]);
          rs += p;
          Ps[wid][r + hi * 8][j * 16 + ln] = f2bf(p);
        }
        #pragma unroll
        for (int off = 1; off < 16; off <<= 1)
          rs += __shfl_xor(rs, off, 32);
        lrow[r] += rs;
        mrow[r] = mnew[r];
      }
    }
    __syncthreads();   // Ps visibility (uniform across block)
    if (active) {
      // O += P * V
      #pragma unroll
      for (int ks2 = 0; ks2 < 2; ++ks2) {
        v16bf pf;
        #pragma unroll
        for (int e = 0; e < 16; ++e) {
          int k = ks2 * 32 + (hi << 4) + e;
          pf[e] = Ps[wid][ln][k];
        }
        #pragma unroll
        for (int dt = 0; dt < 8; ++dt) {
          v16bf vf = load16bf(&VsT[cur][dt * 16 + ln][ks2 * 32 + hi * 16]);
          o[dt] = wmma_bf16(pf, vf, o[dt]);
        }
      }
    }
  }

  // epilogue
  bf16* Ob = O + ((size_t)b * kT + qw + hi * 8) * kHD + h * kD + ln;
  #pragma unroll
  for (int r = 0; r < 8; ++r) {
    float inv = 1.0f / lrow[r];
    #pragma unroll
    for (int dt = 0; dt < 8; ++dt)
      Ob[(size_t)r * kHD + dt * 16] = f2bf(o[dt][r] * inv);
  }
}

// -------------------------------------------------------------------------
extern "C" void kernel_launch(void* const* d_in, const int* in_sizes, int n_in,
                              void* d_out, int out_size, void* d_ws, size_t ws_size,
                              hipStream_t stream) {
  const float* X  = (const float*)d_in[0];
  const float* Wq = (const float*)d_in[1];
  const float* bq = (const float*)d_in[2];
  const float* Wk = (const float*)d_in[3];
  const float* bk = (const float*)d_in[4];
  const float* Wv = (const float*)d_in[5];
  const float* bv = (const float*)d_in[6];
  const float* Wp = (const float*)d_in[7];
  const float* bp = (const float*)d_in[8];
  float* out = (float*)d_out;

  const float qscale = 0.08838834764831845f;   // 1/sqrt(128)

  const size_t headElems = (size_t)kB * kH * kT * kD;  // 8,388,608
  bf16* Qw  = (bf16*)d_ws;
  bf16* Kw  = Qw + headElems;
  bf16* Vw  = Kw + headElems;          // [b,h,d,t]
  bf16* Ow  = Vw + headElems;          // [B*T, H*D]
  bf16* Xb  = Ow + headElems;          // [M, E] bf16
  bf16* WqT = Xb + (size_t)kM * kE;    // [HD, E]
  bf16* WkT = WqT + (size_t)kHD * kE;
  bf16* WvT = WkT + (size_t)kHD * kE;
  bf16* WpT = WvT + (size_t)kHD * kE;  // [E, HD]

  // one-time conversions / transposes (bf16 feeds the async DMA path)
  cvt_x_kernel<<<(kM * kE) / 1024, 256, 0, stream>>>(X, Xb);
  cvt_wT_kernel<<<dim3(kHD / 1024, kE), 256, 0, stream>>>(Wq, WqT, kE, kHD, qscale);
  cvt_wT_kernel<<<dim3(kHD / 1024, kE), 256, 0, stream>>>(Wk, WkT, kE, kHD, 1.0f);
  cvt_wT_kernel<<<dim3(kHD / 1024, kE), 256, 0, stream>>>(Wv, WvT, kE, kHD, 1.0f);
  cvt_wT_kernel<<<dim3(kE / 1024, kHD), 256, 0, stream>>>(Wp, WpT, kHD, kE, 1.0f);

  dim3 gProj(kHD / 128, kM / 64);
  gemm_bf16_kernel<0, kE><<<gProj, 256, 0, stream>>>(Xb, WqT, bq, qscale, Qw);
  gemm_bf16_kernel<0, kE><<<gProj, 256, 0, stream>>>(Xb, WkT, bk, 1.0f, Kw);
  gemm_bf16_kernel<1, kE><<<gProj, 256, 0, stream>>>(Xb, WvT, bv, 1.0f, Vw);

  attn_kernel<<<dim3(kT / 128, kH, kB), 256, 0, stream>>>(Qw, Kw, Vw, Ow);

  out_proj_kernel<<<dim3(kE / 128, kM / 64), 256, 0, stream>>>(Ow, WpT, bp, out);
}
